// GATEncoder_15393162788898
// MI455X (gfx1250) — matrix operand order repaired and checked by reading, not compiled, not run on hardware
//
#include <hip/hip_runtime.h>
#include <math.h>

#define N_NODES 50000
#define NE0     600000
#define NE_TOT  650000
#define F       128
#define NG      64

typedef __attribute__((ext_vector_type(16))) __bf16 v16bf;
typedef __attribute__((ext_vector_type(8)))  __bf16 v8bf;
typedef __attribute__((ext_vector_type(8)))  float  v8f;
typedef unsigned short bfraw;

__device__ __forceinline__ bfraw f2bf(float x) {
  __bf16 h = (__bf16)x;
  return *reinterpret_cast<bfraw*>(&h);
}

// ---------------- utility kernels ----------------

__global__ void zero_f32_kernel(float* __restrict__ p, int n) {
  int i = blockIdx.x * blockDim.x + threadIdx.x;
  if (i < n) p[i] = 0.0f;
}

__global__ void to_bf16_kernel(const float* __restrict__ in, bfraw* __restrict__ out, int n) {
  int i = blockIdx.x * blockDim.x + threadIdx.x;
  if (i < n) out[i] = f2bf(in[i]);
}

// Wt[n*F + k] = W[k*F + n]  (transpose W so B-fragment K-runs are contiguous)
__global__ void wtrans_bf16_kernel(const float* __restrict__ W, bfraw* __restrict__ Wt) {
  int i = blockIdx.x * blockDim.x + threadIdx.x;
  if (i >= F * F) return;
  int n = i >> 7, k = i & 127;
  Wt[i] = f2bf(W[k * F + n]);
}

// ---------------- GEMM: H[50000x128] = A[50000x128] * W[128x128] (bf16 WMMA, f32 acc) ----------------
// grid.x = 3125 (16 rows per block), 256 threads = 8 waves, wave w -> columns [16w,16w+16)

__global__ void __launch_bounds__(256) gemm_bf16_wmma_kernel(
    const bfraw* __restrict__ Araw, const bfraw* __restrict__ Btraw,
    float* __restrict__ H) {
  const __bf16* A  = reinterpret_cast<const __bf16*>(Araw);
  const __bf16* Bt = reinterpret_cast<const __bf16*>(Btraw);
  int wave = threadIdx.x >> 5;
  int lane = threadIdx.x & 31;
  int half = lane >> 4;     // K-half selector per ISA 16-bit layouts
  int lc   = lane & 15;     // row (A) / column (B,C)
  int r0 = blockIdx.x << 4;
  int c0 = wave << 4;
  const __bf16* arow = A  + (size_t)(r0 + lc) * F;
  const __bf16* bcol = Bt + (size_t)(c0 + lc) * F;
  v8f acc = {};
#pragma unroll
  for (int kt = 0; kt < 4; ++kt) {
    int kb = kt << 5;  // K tile base, 32 per WMMA
    // A 16x32 bf16 layout: elems 0..7 -> K = kb+half*8+0..7 ; elems 8..15 -> K = kb+half*8+16..23
    v8bf a0 = *reinterpret_cast<const v8bf*>(arow + kb + half * 8);
    v8bf a1 = *reinterpret_cast<const v8bf*>(arow + kb + half * 8 + 16);
    // B 32x16 bf16 layout: elems 0..7 -> K = kb+half*8+0..7 ; elems 8..15 -> K = kb+half*8+16..23
    v8bf b0 = *reinterpret_cast<const v8bf*>(bcol + kb + half * 8);
    v8bf b1 = *reinterpret_cast<const v8bf*>(bcol + kb + half * 8 + 16);
    v16bf a, b;
#pragma unroll
    for (int i = 0; i < 8; ++i) {
      a[i] = a0[i]; a[i + 8] = a1[i];
      b[i] = b0[i]; b[i + 8] = b1[i];
    }
    acc = __builtin_amdgcn_wmma_f32_16x16x32_bf16(false, a, false, b,
                                                  (short)0, acc, false, false);
  }
  // C/D 16x16 f32 layout: VGPR v, lanes 0-15: M=v ; lanes 16-31: M=v+8 ; N=lane&15
#pragma unroll
  for (int v = 0; v < 8; ++v) {
    H[(size_t)(r0 + v + half * 8) * F + c0 + lc] = acc[v];
  }
}

// ---------------- alpha_s / alpha_d : one wave per node ----------------

__global__ void alpha_kernel(const float* __restrict__ H,
                             const float* __restrict__ a_src,
                             const float* __restrict__ a_dst,
                             float* __restrict__ AS, float* __restrict__ AD) {
  int t = blockIdx.x * blockDim.x + threadIdx.x;
  int node = t >> 5, lane = t & 31;
  if (node >= N_NODES) return;
  const float4* row = reinterpret_cast<const float4*>(H + (size_t)node * F);
  const float4* s4  = reinterpret_cast<const float4*>(a_src);
  const float4* d4  = reinterpret_cast<const float4*>(a_dst);
  float4 v = row[lane];
  float4 s = s4[lane];
  float4 d = d4[lane];
  float ps = v.x * s.x + v.y * s.y + v.z * s.z + v.w * s.w;
  float pd = v.x * d.x + v.y * d.y + v.z * d.z + v.w * d.w;
#pragma unroll
  for (int off = 16; off >= 1; off >>= 1) {
    ps += __shfl_xor(ps, off, 32);
    pd += __shfl_xor(pd, off, 32);
  }
  if (lane == 0) { AS[node] = ps; AD[node] = pd; }
}

// ---------------- edge softmax passes ----------------

__device__ __forceinline__ void edge_endpoints(const int* __restrict__ ei, int e,
                                               int& s, int& d) {
  if (e < NE0) { s = ei[e]; d = ei[NE0 + e]; }
  else         { s = e - NE0; d = e - NE0; }          // appended self-loops
}

// pass 1: leaky-relu logit + encoded float atomic max per destination
__global__ void edge_logits_kernel(const int* __restrict__ ei,
                                   const float* __restrict__ AS,
                                   const float* __restrict__ AD,
                                   float* __restrict__ ebuf,
                                   unsigned* __restrict__ menc) {
  int e = blockIdx.x * blockDim.x + threadIdx.x;
  if (e >= NE_TOT) return;
  int s, d; edge_endpoints(ei, e, s, d);
  float v = AS[s] + AD[d];
  v = v > 0.0f ? v : 0.2f * v;
  ebuf[e] = v;
  unsigned u = __float_as_uint(v);
  u = (u & 0x80000000u) ? ~u : (u | 0x80000000u);   // order-preserving encode
  atomicMax(&menc[d], u);
}

// pass 2: ee = exp(e - max) ; denominator accumulation
__global__ void edge_exp_kernel(const int* __restrict__ ei,
                                float* __restrict__ ebuf,
                                const unsigned* __restrict__ menc,
                                float* __restrict__ sden) {
  int e = blockIdx.x * blockDim.x + threadIdx.x;
  if (e >= NE_TOT) return;
  int s, d; edge_endpoints(ei, e, s, d);
  unsigned u = menc[d];
  float m = 0.0f;
  if (u != 0u) {
    u = (u & 0x80000000u) ? (u & 0x7FFFFFFFu) : ~u;  // decode
    m = __uint_as_float(u);
  }
  float ee = expf(ebuf[e] - m);
  ebuf[e] = ee;
  atomicAdd(&sden[d], ee);
}

// pass 3: out[dst] += coef * h[src]  — one wave per edge, float4 per lane
__global__ void edge_aggregate_kernel(const int* __restrict__ ei,
                                      const float* __restrict__ ebuf,
                                      const float* __restrict__ sden,
                                      const float* __restrict__ H,
                                      float* __restrict__ out) {
  int t = blockIdx.x * blockDim.x + threadIdx.x;
  int e = t >> 5, lane = t & 31;
  if (e >= NE_TOT) return;
  int s, d; edge_endpoints(ei, e, s, d);
  float coef = ebuf[e] / (sden[d] + 1e-16f);
  const float4* hr = reinterpret_cast<const float4*>(H + (size_t)s * F);
  float4 v = hr[lane];
  float* od = out + (size_t)d * F + lane * 4;
  atomicAdd(od + 0, v.x * coef);
  atomicAdd(od + 1, v.y * coef);
  atomicAdd(od + 2, v.z * coef);
  atomicAdd(od + 3, v.w * coef);
}

// ---------------- bias + ELU (in place) ----------------

__global__ void bias_elu_kernel(float* __restrict__ out, const float* __restrict__ b, int n) {
  int i = blockIdx.x * blockDim.x + threadIdx.x;
  if (i >= n) return;
  float v = out[i] + b[i & 127];
  out[i] = v > 0.0f ? v : expm1f(v);
}

// ---------------- global mean pool ----------------

__global__ void pool_accum_kernel(const float* __restrict__ Hf,
                                  const int* __restrict__ batch,
                                  float* __restrict__ pool,
                                  float* __restrict__ cnt) {
  int t = blockIdx.x * blockDim.x + threadIdx.x;
  int node = t >> 5, lane = t & 31;
  if (node >= N_NODES) return;
  int g = batch[node];
  const float4* row = reinterpret_cast<const float4*>(Hf + (size_t)node * F);
  float4 v = row[lane];
  float* pg = pool + (size_t)g * F + lane * 4;
  atomicAdd(pg + 0, v.x);
  atomicAdd(pg + 1, v.y);
  atomicAdd(pg + 2, v.z);
  atomicAdd(pg + 3, v.w);
  if (lane == 0) atomicAdd(&cnt[g], 1.0f);
}

__global__ void pool_final_kernel(const float* __restrict__ pool,
                                  const float* __restrict__ cnt,
                                  float* __restrict__ out) {
  int i = blockIdx.x * blockDim.x + threadIdx.x;
  if (i >= NG * F) return;
  out[i] = pool[i] / fmaxf(cnt[i >> 7], 1.0f);
}

// ---------------- host launcher ----------------

extern "C" void kernel_launch(void* const* d_in, const int* in_sizes, int n_in,
                              void* d_out, int out_size, void* d_ws, size_t ws_size,
                              hipStream_t stream) {
  (void)in_sizes; (void)n_in; (void)out_size; (void)ws_size;
  const float* x     = (const float*)d_in[0];
  const int*   ei    = (const int*)d_in[1];
  const int*   batch = (const int*)d_in[2];

  char* w = (char*)d_ws;
  auto alloc = [&](size_t bytes) -> void* {
    void* p = (void*)w;
    w += (bytes + 255) & ~(size_t)255;
    return p;
  };
  bfraw*    Abf  = (bfraw*)alloc((size_t)N_NODES * F * 2);
  bfraw*    Wt   = (bfraw*)alloc((size_t)F * F * 2);
  float*    H    = (float*)alloc((size_t)N_NODES * F * 4);
  float*    OUT  = (float*)alloc((size_t)N_NODES * F * 4);
  float*    AS   = (float*)alloc((size_t)N_NODES * 4);
  float*    AD   = (float*)alloc((size_t)N_NODES * 4);
  unsigned* MENC = (unsigned*)alloc((size_t)N_NODES * 4);
  float*    SDEN = (float*)alloc((size_t)N_NODES * 4);
  float*    EBUF = (float*)alloc((size_t)NE_TOT * 4);
  float*    POOL = (float*)alloc((size_t)NG * F * 4);
  float*    CNT  = (float*)alloc((size_t)NG * 4);

  const int T = 256;
  auto cdiv = [](int a, int b) { return (a + b - 1) / b; };

  const float* xin = x;
  for (int L = 0; L < 3; ++L) {
    const float* Wl   = (const float*)d_in[3 + 4 * L];
    const float* asrc = (const float*)d_in[4 + 4 * L];
    const float* adst = (const float*)d_in[5 + 4 * L];
    const float* bl   = (const float*)d_in[6 + 4 * L];

    to_bf16_kernel<<<cdiv(N_NODES * F, T), T, 0, stream>>>(xin, Abf, N_NODES * F);
    wtrans_bf16_kernel<<<cdiv(F * F, T), T, 0, stream>>>(Wl, Wt);
    gemm_bf16_wmma_kernel<<<N_NODES / 16, T, 0, stream>>>(Abf, Wt, H);
    alpha_kernel<<<cdiv(N_NODES * 32, T), T, 0, stream>>>(H, asrc, adst, AS, AD);
    zero_f32_kernel<<<cdiv(N_NODES * F, T), T, 0, stream>>>(OUT, N_NODES * F);
    zero_f32_kernel<<<cdiv(N_NODES, T), T, 0, stream>>>((float*)MENC, N_NODES);
    zero_f32_kernel<<<cdiv(N_NODES, T), T, 0, stream>>>(SDEN, N_NODES);
    edge_logits_kernel<<<cdiv(NE_TOT, T), T, 0, stream>>>(ei, AS, AD, EBUF, MENC);
    edge_exp_kernel<<<cdiv(NE_TOT, T), T, 0, stream>>>(ei, EBUF, MENC, SDEN);
    edge_aggregate_kernel<<<cdiv(NE_TOT * 32, T), T, 0, stream>>>(ei, EBUF, SDEN, H, OUT);
    bias_elu_kernel<<<cdiv(N_NODES * F, T), T, 0, stream>>>(OUT, bl, N_NODES * F);
    xin = OUT;
  }

  zero_f32_kernel<<<cdiv(NG * F, T), T, 0, stream>>>(POOL, NG * F);
  zero_f32_kernel<<<1, T, 0, stream>>>(CNT, NG);
  pool_accum_kernel<<<cdiv(N_NODES * 32, T), T, 0, stream>>>(OUT, batch, POOL, CNT);
  pool_final_kernel<<<cdiv(NG * F, T), T, 0, stream>>>(POOL, CNT, (float*)d_out);
}
